// PredictbyLSTM_86182813761599
// MI455X (gfx1250) — matrix operand order repaired and checked by reading, not compiled, not run on hardware
//
#include <hip/hip_runtime.h>
#include <hip/hip_bf16.h>

// ---------------------------------------------------------------------------
// LSTM-style persistent kernel for MI455X (gfx1250, wave32, WMMA).
//   z = h @ (W_ih+W_hh)^T + (b_ih+b_hh)     (GEMM1, K=256, N=2048)
//   i,f,g,o = gates(z); cc = f*cc + i*g
//   h = (o*tanh(cc)) @ W_hr^T               (GEMM2, K=512, N=256)
// Batch rows are independent recurrences -> 16-row batch tile per workgroup,
// h/cc resident, pre-swizzled bf16 weight fragments streamed from L2 every
// step, all matmuls on v_wmma_f32_16x16x32_bf16.
//
// Round-2: asm barrier stops LICM from hoisting 160 t-invariant weight
//          fragments into a 1280-VGPR spill.
// Round-3: launder the pointer into an explicit address_space(1) pointer so
//          weight streams use global_load_b128 (LOADcnt only) instead of
//          flat_load_b128 (LOADcnt+DScnt, ties up the LDS path), and fold
//          wave/lane into one base so every fragment is SGPR base + lane*32B
//          VGPR offset + constant ioffset.
// ---------------------------------------------------------------------------

typedef __attribute__((ext_vector_type(16))) __bf16 v16bf;
typedef __attribute__((ext_vector_type(8)))  __bf16 v8bf;
typedef __attribute__((ext_vector_type(8)))  float  v8f;

typedef __attribute__((address_space(1))) const __bf16  gbf16;
typedef __attribute__((address_space(1))) const v16bf   gv16bf;

#define B_SZ  2048
#define H_SZ  512
#define D_SZ  256
#define NSTEP 64
#define G4H   2048   // 4*H

// LDS row strides (bf16 elements). Requirements:
//  - stride*2 bytes multiple of 16 (legal ds_load_b128 alignment)
//  - stride*2 % 256 == 48 -> 16 rows map to 16 distinct bank groups
#define HSTR 280     // >= 256 (h tile: 16 x D)
#define SSTR 536     // >= 512 (s tile: 16 x H)

__device__ __forceinline__ float fast_sigmoid(float x) {
  return 1.0f / (1.0f + __expf(-x));
}
__device__ __forceinline__ float fast_tanh(float x) {
  // 2*sigmoid(2x)-1 : overflow-safe, single v_exp_f32
  return 2.0f / (1.0f + __expf(-2.0f * x)) - 1.0f;
}

// --- Prep: combine input/hidden weights into bf16 WMMA B-fragment order -----
// B-matrix for GEMM1 is Wc^T (K=256 x N=2048). Fragment tile (ntile, ktile):
// lane L holds 16 contiguous bf16 with  N = ntile*16 + (L%16),
//                                       K = ktile*32 + (L/16)*16 + j.
__global__ void prep_w1(const float* __restrict__ Wih,
                        const float* __restrict__ Whh,
                        __bf16* __restrict__ Wf1) {
  int gid = blockIdx.x * blockDim.x + threadIdx.x;
  if (gid >= G4H * D_SZ) return;
  int t   = gid >> 9;        // fragment tile = ntile*8 + ktile (512 elems each)
  int rem = gid & 511;
  int L = rem >> 4, j = rem & 15;
  int ntile = t >> 3, ktile = t & 7;
  int n = ntile * 16 + (L & 15);
  int k = ktile * 32 + (L >> 4) * 16 + j;
  float w = Wih[n * D_SZ + k] + Whh[n * D_SZ + k];
  Wf1[gid] = (__bf16)w;
}

// Same swizzle for W_hr^T (K=512 x N=256): tile = ntile2*16 + ktile2.
__global__ void prep_w2(const float* __restrict__ Whr,
                        __bf16* __restrict__ Wf2) {
  int gid = blockIdx.x * blockDim.x + threadIdx.x;
  if (gid >= D_SZ * H_SZ) return;
  int t   = gid >> 9;
  int rem = gid & 511;
  int L = rem >> 4, j = rem & 15;
  int ntile = t >> 4, ktile = t & 15;
  int n = ntile * 16 + (L & 15);          // < 256
  int k = ktile * 32 + (L >> 4) * 16 + j; // < 512
  Wf2[gid] = (__bf16)Whr[n * H_SZ + k];
}

__global__ void prep_bias(const float* __restrict__ bih,
                          const float* __restrict__ bhh,
                          float* __restrict__ bc) {
  int gid = blockIdx.x * blockDim.x + threadIdx.x;
  if (gid < G4H) bc[gid] = bih[gid] + bhh[gid];
}

// --- Persistent recurrence kernel ------------------------------------------
// grid = B/16 blocks, 256 threads (8 wave32). Wave w owns H-slice
// [w*64, w*64+64): the matching i/f/g/o columns, so gate fusion and the cc
// update stay in this wave's accumulator registers.
__global__ __launch_bounds__(256, 1)
void lstm_persist(const float* __restrict__ cinit,
                  const __bf16* __restrict__ Wf1,
                  const __bf16* __restrict__ Wf2,
                  const float* __restrict__ bias,
                  float* __restrict__ out) {
  __shared__ __bf16 h_tile[16 * HSTR];   // 16 x 256 (padded) bf16
  __shared__ __bf16 s_tile[16 * SSTR];   // 16 x 512 (padded) bf16

  const int tid   = threadIdx.x;
  const int wave  = tid >> 5;
  const int lane  = tid & 31;            // wave32 (gfx1250)
  const int lhalf = lane >> 4;
  const int l16   = lane & 15;
  const int b0    = blockIdx.x * 16;

  // h0 = 0
  for (int i = tid; i < 16 * HSTR; i += 256) h_tile[i] = (__bf16)0.0f;

  // cc resident in C-fragment layout: frag tt covers cols wave*64+tt*16+l16,
  // rows v + lhalf*8 (matches 16x16 f32 C/D VGPR layout).
  v8f cc[4];
#pragma unroll
  for (int tt = 0; tt < 4; ++tt) {
    const int n = wave * 64 + tt * 16 + l16;
#pragma unroll
    for (int v = 0; v < 8; ++v) {
      const int m = v + lhalf * 8;
      cc[tt][v] = cinit[(size_t)(b0 + m) * H_SZ + n];
    }
  }

  float bias_s[4][4];
#pragma unroll
  for (int g = 0; g < 4; ++g)
#pragma unroll
    for (int tt = 0; tt < 4; ++tt)
      bias_s[g][tt] = bias[g * H_SZ + wave * 64 + tt * 16 + l16];

  __syncthreads();

  for (int t = 0; t < NSTEP; ++t) {
    // Re-materialize weight base addresses each step (blocks LICM hoist ->
    // spill; weights are L2-resident, re-streaming them is the design), then
    // cast to explicit global-AS pointers so loads are global_load_b128 with
    // SGPR base + lane*32B VGPR offset + constant instruction offset.
    unsigned long long w1a = (unsigned long long)Wf1;
    unsigned long long w2a = (unsigned long long)Wf2;
    asm volatile("" : "+s"(w1a), "+s"(w2a));
    // per-wave, per-lane base: fragment idx = ((ntile*8+kt)*32 + lane)*16
    gbf16* base1 = (gbf16*)w1a + ((size_t)(wave * 4) * 8) * 512 + lane * 16;
    gbf16* base2 = (gbf16*)w2a + ((size_t)(wave * 2) * 16) * 512 + lane * 16;

    // ---- Phase 1: GEMM1 + gates + cc update + s -> LDS ----
    // A-fragments of the h tile (16x256, K split into 8 tiles of 32).
    // A fragments in this code's contiguous k order: element j has K = (lane>=16?16:0)+j,
    // i.e. one contiguous 16-element run per lane (matches B fragments).
    v16bf a[8];
#pragma unroll
    for (int kt = 0; kt < 8; ++kt) {
      const __bf16* p = &h_tile[l16 * HSTR + kt * 32 + lhalf * 16];
      v8bf lo = *(const v8bf*)(p);
      v8bf hi = *(const v8bf*)(p + 8);
      a[kt] = __builtin_shufflevector(lo, hi, 0, 1, 2, 3, 4, 5, 6, 7,
                                      8, 9, 10, 11, 12, 13, 14, 15);
    }

#pragma unroll
    for (int tt = 0; tt < 4; ++tt) {
      v8f z[4];
#pragma unroll
      for (int g = 0; g < 4; ++g) {
        v8f acc;
#pragma unroll
        for (int v = 0; v < 8; ++v) acc[v] = bias_s[g][tt];
#pragma unroll
        for (int kt = 0; kt < 8; ++kt) {
          // constant element offset: ((g*32 + tt)*8 + kt) * 512 (rel. base1)
          const v16bf bf =
              *(gv16bf*)(base1 + ((size_t)(g * 32 + tt) * 8 + kt) * 512);
          acc = __builtin_amdgcn_wmma_f32_16x16x32_bf16(
              false, a[kt], false, bf, (short)0, acc, false, false);
        }
        z[g] = acc;
      }
      // fuse gates; trans ops co-execute with WMMA on gfx1250
#pragma unroll
      for (int v = 0; v < 8; ++v) {
        float i_ = fast_sigmoid(z[0][v]);
        float f_ = fast_sigmoid(z[1][v]);
        float g_ = fast_tanh(z[2][v]);
        float o_ = fast_sigmoid(z[3][v]);
        float nc = f_ * cc[tt][v] + i_ * g_;
        cc[tt][v] = nc;
        const int m = v + lhalf * 8;
        const int n = wave * 64 + tt * 16 + l16;
        s_tile[m * SSTR + n] = (__bf16)(o_ * fast_tanh(nc));
      }
    }

    __syncthreads();  // s complete; everyone done reading h

    // ---- Phase 2: GEMM2  h_new = S @ W_hr^T  (K=512, wave owns 32 cols) ----
    v8f hn[2];
#pragma unroll
    for (int nt = 0; nt < 2; ++nt)
#pragma unroll
      for (int v = 0; v < 8; ++v) hn[nt][v] = 0.0f;

#pragma unroll
    for (int kt = 0; kt < 16; ++kt) {
      const __bf16* p = &s_tile[l16 * SSTR + kt * 32 + lhalf * 16];
      v8bf lo = *(const v8bf*)(p);
      v8bf hi = *(const v8bf*)(p + 8);
      v16bf af = __builtin_shufflevector(lo, hi, 0, 1, 2, 3, 4, 5, 6, 7,
                                         8, 9, 10, 11, 12, 13, 14, 15);
#pragma unroll
      for (int nt = 0; nt < 2; ++nt) {
        // constant element offset: (nt*16 + kt) * 512 (rel. base2)
        const v16bf bf = *(gv16bf*)(base2 + ((size_t)(nt * 16 + kt)) * 512);
        hn[nt] = __builtin_amdgcn_wmma_f32_16x16x32_bf16(
            false, af, false, bf, (short)0, hn[nt], false, false);
      }
    }

    // write output (B, D, STEP) and refresh h tile for next step
#pragma unroll
    for (int nt = 0; nt < 2; ++nt) {
#pragma unroll
      for (int v = 0; v < 8; ++v) {
        const int m = v + lhalf * 8;
        const int n = wave * 32 + nt * 16 + l16;
        out[(size_t)(b0 + m) * (D_SZ * NSTEP) + (size_t)n * NSTEP + t] =
            hn[nt][v];
        h_tile[m * HSTR + n] = (__bf16)hn[nt][v];
      }
    }

    __syncthreads();  // h tile ready for next step's GEMM1
  }
}

// ---------------------------------------------------------------------------
extern "C" void kernel_launch(void* const* d_in, const int* in_sizes, int n_in,
                              void* d_out, int out_size, void* d_ws,
                              size_t ws_size, hipStream_t stream) {
  const float* c    = (const float*)d_in[0];  // (B, H, 1)
  const float* W_ih = (const float*)d_in[1];  // (4H, D)
  const float* W_hh = (const float*)d_in[2];  // (4H, D)
  const float* b_ih = (const float*)d_in[3];  // (4H,)
  const float* b_hh = (const float*)d_in[4];  // (4H,)
  const float* W_hr = (const float*)d_in[5];  // (D, H)
  float* out = (float*)d_out;                 // (B, D, STEP)

  char* ws = (char*)d_ws;
  __bf16* Wf1 = (__bf16*)ws;                               // 1 MB
  __bf16* Wf2 = (__bf16*)(ws + (1u << 20));                // 256 KB
  float* bc   = (float*)(ws + (1u << 20) + (256u << 10));  // 8 KB

  prep_w1<<<(G4H * D_SZ + 255) / 256, 256, 0, stream>>>(W_ih, W_hh, Wf1);
  prep_w2<<<(D_SZ * H_SZ + 255) / 256, 256, 0, stream>>>(W_hr, Wf2);
  prep_bias<<<(G4H + 255) / 256, 256, 0, stream>>>(b_ih, b_hh, bc);

  lstm_persist<<<B_SZ / 16, 256, 0, stream>>>(c, Wf1, Wf2, bc, out);
}
